// MultiHeadAttention_42271068127395
// MI455X (gfx1250) — matrix operand order repaired and checked by reading, not compiled, not run on hardware
//
#include <hip/hip_runtime.h>
#include <hip/hip_bf16.h>

// ---------------------------------------------------------------------------
// LoRA multi-head attention forward, MI455X (gfx1250, wave32, WMMA).
// All GEMM-shaped work runs through v_wmma_f32_16x16x32_bf16 (fp32 accum).
// ---------------------------------------------------------------------------

#define D_MODEL   1024
#define NUM_HEADS 16
#define D_K       64
#define SEQ_T     2048
#define BATCH     2
#define NROW      (BATCH * SEQ_T)   // 4096
#define R_LORA    8
#define LORA_SCALE 2.0f             // alpha / r = 16 / 8

typedef __bf16        bf16x16 __attribute__((ext_vector_type(16)));
typedef float         f32x8   __attribute__((ext_vector_type(8)));
typedef unsigned int  u32x4   __attribute__((ext_vector_type(4)));

union Frag16 {                       // one WMMA 16-bit operand (8 VGPRs/lane)
  bf16x16        v;
  unsigned short u[16];
  u32x4          q[2];
};

__device__ __forceinline__ unsigned short f32_to_bf16_rne(float f) {
  union { float f; unsigned u; } c; c.f = f;
  unsigned r = c.u + 0x7FFFu + ((c.u >> 16) & 1u);  // round-to-nearest-even
  return (unsigned short)(r >> 16);
}

// ---------------------------------------------------------------------------
// Stage 1: fp32 -> bf16 conversion (inputs + weights)
// ---------------------------------------------------------------------------
__global__ void cvt_f32_to_bf16(const float* __restrict__ src,
                                unsigned short* __restrict__ dst, int n) {
  int i = blockIdx.x * blockDim.x + threadIdx.x;
  if (i < n) dst[i] = f32_to_bf16_rne(src[i]);
}

// ---------------------------------------------------------------------------
// Stage 2: LoRA left factor  Xa[row][r] = sum_d x[row][d] * A[r][d]
// One wave per row (fp32 inputs, cross-lane shfl reduction).
// ---------------------------------------------------------------------------
__global__ __launch_bounds__(256) void lora_xa_kernel(
    const float* __restrict__ x, const float* __restrict__ A,
    float* __restrict__ xa) {
  const int wave = threadIdx.x >> 5;
  const int lane = threadIdx.x & 31;
  const int row  = blockIdx.x * 8 + wave;
  const float* xr = x + (size_t)row * D_MODEL;
  #pragma unroll
  for (int r = 0; r < R_LORA; ++r) {
    const float* ar = A + r * D_MODEL;
    float s = 0.f;
    for (int d = lane; d < D_MODEL; d += 32) s += xr[d] * ar[d];
    #pragma unroll
    for (int m = 16; m >= 1; m >>= 1) s += __shfl_xor(s, m, 32);
    if (lane == 0) xa[row * R_LORA + r] = s;
  }
}

// ---------------------------------------------------------------------------
// Stage 3/5: WMMA GEMM  out = Abf(4096x1024) @ Wbf^T(1024x1024) + bias
//                            (+ LORA_SCALE * Xa @ Bm^T  if has_lora)
// Block: 256 threads = 8 waves as 4(M) x 2(N); block tile 128 x 128,
// wave tile 32 x 64: two M-subtiles share every B fragment, so each K-step
// is 12 b128 loads for 8 WMMAs; K-loop unrolled x2 for deeper pipelining.
// out_mode: 0 = fp32 [NROW,D];  1 = bf16 [B,H,T,64];  2 = bf16 [B,H,64,T].
// ---------------------------------------------------------------------------
__global__ __launch_bounds__(256) void gemm_wmma_kernel(
    const unsigned short* __restrict__ Abf,
    const unsigned short* __restrict__ Wbf,
    const float* __restrict__ bias,
    const float* __restrict__ Xa,
    const float* __restrict__ Bm,
    float* __restrict__ outf,
    unsigned short* __restrict__ outb,
    int out_mode, int has_lora) {
  const int lane = threadIdx.x & 31;
  const int w    = threadIdx.x >> 5;
  const int wm   = w & 3, wn = w >> 2;
  const int row_base = blockIdx.x * 128 + wm * 32;
  const int col_base = blockIdx.y * 128 + wn * 64;
  const int lm  = lane & 15;
  const int lhi = lane >> 4;

  f32x8 acc[2][4];
  #pragma unroll
  for (int ms = 0; ms < 2; ++ms)
    #pragma unroll
    for (int i = 0; i < 4; ++i)
      #pragma unroll
      for (int e = 0; e < 8; ++e) acc[ms][i][e] = 0.f;

  const unsigned short* aptr0 = Abf + (size_t)(row_base + lm) * D_MODEL;
  const unsigned short* aptr1 = aptr0 + (size_t)16 * D_MODEL;

  #pragma unroll 2
  for (int k0 = 0; k0 < D_MODEL; k0 += 32) {
    Frag16 a0, a1;                    // A: lane=row, K = 16*lhi + e
    const int kb = k0 + lhi * 16;
    a0.q[0] = *(const u32x4*)(aptr0 + kb);
    a0.q[1] = *(const u32x4*)(aptr0 + kb + 8);
    a1.q[0] = *(const u32x4*)(aptr1 + kb);
    a1.q[1] = *(const u32x4*)(aptr1 + kb + 8);
    // Speculative prefetch: out-of-range prefetches are silently dropped,
    // so no guard branch in the inner loop.
    __builtin_prefetch(aptr0 + kb + 64, 0, 1);
    __builtin_prefetch(aptr1 + kb + 64, 0, 1);
    #pragma unroll
    for (int nt = 0; nt < 4; ++nt) {  // B: lane=col, K = 16*lhi + e
      Frag16 b;
      const int ocol = col_base + nt * 16 + lm;
      const unsigned short* bptr =
          Wbf + (size_t)ocol * D_MODEL + k0 + lhi * 16;
      b.q[0] = *(const u32x4*)(bptr);
      b.q[1] = *(const u32x4*)(bptr + 8);
      acc[0][nt] = __builtin_amdgcn_wmma_f32_16x16x32_bf16(
          false, a0.v, false, b.v, (short)0, acc[0][nt], false, false);
      acc[1][nt] = __builtin_amdgcn_wmma_f32_16x16x32_bf16(
          false, a1.v, false, b.v, (short)0, acc[1][nt], false, false);
    }
  }

  // Epilogue: bias + LoRA rank-8 outer product, then scatter per layout.
  float bmv[4][8], bsv[4];
  #pragma unroll
  for (int nt = 0; nt < 4; ++nt) {
    const int col = col_base + nt * 16 + lm;
    bsv[nt] = bias[col];
    if (has_lora) {
      const float* bp = Bm + col * R_LORA;
      #pragma unroll
      for (int r = 0; r < R_LORA; ++r) bmv[nt][r] = bp[r];
    }
  }
  #pragma unroll
  for (int ms = 0; ms < 2; ++ms) {
    #pragma unroll
    for (int v = 0; v < 8; ++v) {     // C: lane=col, row = v + 8*lhi
      const int row = row_base + ms * 16 + lhi * 8 + v;
      float xr[R_LORA];
      if (has_lora) {
        const float* xp = Xa + row * R_LORA;
        #pragma unroll
        for (int r = 0; r < R_LORA; ++r) xr[r] = xp[r];
      }
      #pragma unroll
      for (int nt = 0; nt < 4; ++nt) {
        const int col = col_base + nt * 16 + lm;
        float val = acc[ms][nt][v] + bsv[nt];
        if (has_lora) {
          float l = 0.f;
          #pragma unroll
          for (int r = 0; r < R_LORA; ++r) l += xr[r] * bmv[nt][r];
          val += LORA_SCALE * l;
        }
        if (out_mode == 0) {
          outf[(size_t)row * D_MODEL + col] = val;
        } else {
          const int bb = row >> 11, t = row & (SEQ_T - 1);
          const int h  = col >> 6,  d = col & (D_K - 1);
          size_t idx = (out_mode == 1)
              ? ((size_t)((bb * NUM_HEADS + h) * SEQ_T + t)) * D_K + d
              : ((size_t)((bb * NUM_HEADS + h) * D_K + d)) * SEQ_T + t;
          outb[idx] = f32_to_bf16_rne(val);
        }
      }
    }
  }
}

// ---------------------------------------------------------------------------
// Stage 4: flash attention.  One wave = one 16-query tile, streams keys 32
// at a time.  S = Q*K^T (4 WMMA), mask+scale, online softmax with shfl_xor
// row reductions, P transposed C-layout -> A-layout through LDS (bf16),
// O += P*V (4 WMMA, V stored [B,H,64,T] so B-frags are contiguous).
// ---------------------------------------------------------------------------
__global__ __launch_bounds__(256) void flash_attn_kernel(
    const unsigned short* __restrict__ Qb,   // [B,H,T,64]
    const unsigned short* __restrict__ Kb,   // [B,H,T,64]
    const unsigned short* __restrict__ Vt,   // [B,H,64,T]
    const int* __restrict__ mask,            // [T,T] (broadcast b,h)
    unsigned short* __restrict__ ctx) {      // [NROW, D_MODEL] bf16
  __shared__ unsigned short ldsP[8][16 * 32];   // per-wave P tile (bf16)

  const int lane = threadIdx.x & 31;
  const int w    = threadIdx.x >> 5;
  const int lm   = lane & 15, lhi = lane >> 4;
  const int bh   = blockIdx.x;                  // 0..31
  const int b    = bh >> 4, h = bh & 15;
  const int q0   = blockIdx.y * 128 + w * 16;   // query tile base

  const unsigned short* Qh = Qb + (size_t)bh * SEQ_T * D_K;
  const unsigned short* Kh = Kb + (size_t)bh * SEQ_T * D_K;
  const unsigned short* Vh = Vt + (size_t)bh * D_K * SEQ_T;

  Frag16 aq[2];                      // Q fragments for d-dim 0..31 / 32..63
  {
    const unsigned short* qp = Qh + (size_t)(q0 + lm) * D_K;
    #pragma unroll
    for (int s = 0; s < 2; ++s) {
      const int kb = s * 32 + lhi * 16;
      aq[s].q[0] = *(const u32x4*)(qp + kb);
      aq[s].q[1] = *(const u32x4*)(qp + kb + 8);
    }
  }

  f32x8 ao[4];
  #pragma unroll
  for (int i = 0; i < 4; ++i)
    #pragma unroll
    for (int e = 0; e < 8; ++e) ao[i][e] = 0.f;
  float rmax[8], rsum[8];
  #pragma unroll
  for (int v = 0; v < 8; ++v) { rmax[v] = -3.0e38f; rsum[v] = 0.f; }

  for (int kk = 0; kk < SEQ_T; kk += 32) {
    // ---- S = Q * K^T for 32 keys (two 16-col tiles, two K-steps each) ----
    f32x8 s[2];
    #pragma unroll
    for (int j = 0; j < 2; ++j) {
      f32x8 sj;
      #pragma unroll
      for (int e = 0; e < 8; ++e) sj[e] = 0.f;
      #pragma unroll
      for (int ks = 0; ks < 2; ++ks) {
        Frag16 bk;                   // B[k=d][n=key]: lane=key row, contiguous d
        const unsigned short* kp =
            Kh + (size_t)(kk + j * 16 + lm) * D_K + ks * 32 + lhi * 16;
        bk.q[0] = *(const u32x4*)(kp);
        bk.q[1] = *(const u32x4*)(kp + 8);
        sj = __builtin_amdgcn_wmma_f32_16x16x32_bf16(
            false, aq[ks].v, false, bk.v, (short)0, sj, false, false);
      }
      s[j] = sj;
    }

    // ---- scale by 1/sqrt(d_k) and apply mask (mask lives in L2) ----
    #pragma unroll
    for (int j = 0; j < 2; ++j) {
      const int tk = kk + j * 16 + lm;
      #pragma unroll
      for (int v = 0; v < 8; ++v) {
        const int tq = q0 + lhi * 8 + v;
        float val = s[j][v] * 0.125f;
        if (mask[(size_t)tq * SEQ_T + tk] == 0) val = -1.0e9f;
        s[j][v] = val;
      }
    }

    // ---- online softmax: row max / exp / row sum ----
    float pm[8];
    #pragma unroll
    for (int v = 0; v < 8; ++v) pm[v] = fmaxf(s[0][v], s[1][v]);
    #pragma unroll
    for (int m = 8; m >= 1; m >>= 1)
      #pragma unroll
      for (int v = 0; v < 8; ++v)
        pm[v] = fmaxf(pm[v], __shfl_xor(pm[v], m, 32));

    float scale_old[8], ps[8];
    #pragma unroll
    for (int v = 0; v < 8; ++v) {
      const float nm = fmaxf(rmax[v], pm[v]);
      scale_old[v] = __expf(rmax[v] - nm);
      rmax[v] = nm;
      const float p0 = __expf(s[0][v] - nm);
      const float p1 = __expf(s[1][v] - nm);
      s[0][v] = p0; s[1][v] = p1;
      ps[v] = p0 + p1;
    }
    #pragma unroll
    for (int m = 8; m >= 1; m >>= 1)
      #pragma unroll
      for (int v = 0; v < 8; ++v) ps[v] += __shfl_xor(ps[v], m, 32);
    #pragma unroll
    for (int v = 0; v < 8; ++v) rsum[v] = rsum[v] * scale_old[v] + ps[v];
    #pragma unroll
    for (int nt = 0; nt < 4; ++nt)
      #pragma unroll
      for (int v = 0; v < 8; ++v) ao[nt][v] *= scale_old[v];

    // ---- transpose P (C-layout f32, lane=col) -> A-layout bf16 via LDS ----
    unsigned short* lp = ldsP[w];
    #pragma unroll
    for (int j = 0; j < 2; ++j)
      #pragma unroll
      for (int v = 0; v < 8; ++v)
        lp[(lhi * 8 + v) * 32 + j * 16 + lm] = f32_to_bf16_rne(s[j][v]);
    asm volatile("s_wait_dscnt 0" ::: "memory");   // wave-local RAW on LDS
    Frag16 ap;
    ap.q[0] = *(const u32x4*)(lp + lm * 32 + lhi * 16);
    ap.q[1] = *(const u32x4*)(lp + lm * 32 + lhi * 16 + 8);

    // ---- O += P * V ----
    #pragma unroll
    for (int nt = 0; nt < 4; ++nt) {
      Frag16 bv;                     // B[k=key][n=d]: lane=d, contiguous key
      const unsigned short* vp =
          Vh + (size_t)(nt * 16 + lm) * SEQ_T + kk + lhi * 16;
      bv.q[0] = *(const u32x4*)(vp);
      bv.q[1] = *(const u32x4*)(vp + 8);
      ao[nt] = __builtin_amdgcn_wmma_f32_16x16x32_bf16(
          false, ap.v, false, bv.v, (short)0, ao[nt], false, false);
    }
  }

  // ---- normalize and store context [B,T,D_MODEL] bf16 ----
  float rinv[8];
  #pragma unroll
  for (int v = 0; v < 8; ++v) rinv[v] = 1.f / rsum[v];
  #pragma unroll
  for (int nt = 0; nt < 4; ++nt)
    #pragma unroll
    for (int v = 0; v < 8; ++v) {
      const int row = b * SEQ_T + q0 + lhi * 8 + v;
      const int col = h * D_K + nt * 16 + lm;
      ctx[(size_t)row * D_MODEL + col] = f32_to_bf16_rne(ao[nt][v] * rinv[v]);
    }
}

// ---------------------------------------------------------------------------
// Host launch
// ---------------------------------------------------------------------------
extern "C" void kernel_launch(void* const* d_in, const int* in_sizes, int n_in,
                              void* d_out, int out_size, void* d_ws,
                              size_t ws_size, hipStream_t stream) {
  (void)in_sizes; (void)n_in; (void)out_size; (void)ws_size;
  const float* q    = (const float*)d_in[0];
  const float* k    = (const float*)d_in[1];
  const float* v    = (const float*)d_in[2];
  const int*   mask = (const int*)d_in[3];
  const float* Wq = (const float*)d_in[4],  *bq = (const float*)d_in[5];
  const float* Aq = (const float*)d_in[6],  *Bq = (const float*)d_in[7];
  const float* Wk = (const float*)d_in[8],  *bk = (const float*)d_in[9];
  const float* Ak = (const float*)d_in[10], *Bk = (const float*)d_in[11];
  const float* Wv = (const float*)d_in[12], *bv = (const float*)d_in[13];
  const float* Av = (const float*)d_in[14], *Bv = (const float*)d_in[15];
  const float* Wo = (const float*)d_in[16], *bo = (const float*)d_in[17];
  float* out = (float*)d_out;

  // Workspace layout (bf16 region then fp32 region), total ~64.4 MiB.
  const size_t NX = (size_t)NROW * D_MODEL;    // 4,194,304
  const size_t NW = (size_t)D_MODEL * D_MODEL; // 1,048,576
  unsigned short* wsu = (unsigned short*)d_ws;
  unsigned short* xqb = wsu;
  unsigned short* xkb = xqb + NX;
  unsigned short* xvb = xkb + NX;
  unsigned short* wqb = xvb + NX;
  unsigned short* wkb = wqb + NW;
  unsigned short* wvb = wkb + NW;
  unsigned short* wob = wvb + NW;
  unsigned short* Qb  = wob + NW;
  unsigned short* Kb  = Qb + NX;
  unsigned short* Vtb = Kb + NX;
  unsigned short* ctx = Vtb + NX;
  float* xaf = (float*)(ctx + NX);
  float* xaq = xaf;
  float* xak = xaf + (size_t)NROW * R_LORA;
  float* xav = xak + (size_t)NROW * R_LORA;

  // Stage 1: convert activations + weights to bf16.
  cvt_f32_to_bf16<<<(int)(NX / 256), 256, 0, stream>>>(q, xqb, (int)NX);
  cvt_f32_to_bf16<<<(int)(NX / 256), 256, 0, stream>>>(k, xkb, (int)NX);
  cvt_f32_to_bf16<<<(int)(NX / 256), 256, 0, stream>>>(v, xvb, (int)NX);
  cvt_f32_to_bf16<<<(int)(NW / 256), 256, 0, stream>>>(Wq, wqb, (int)NW);
  cvt_f32_to_bf16<<<(int)(NW / 256), 256, 0, stream>>>(Wk, wkb, (int)NW);
  cvt_f32_to_bf16<<<(int)(NW / 256), 256, 0, stream>>>(Wv, wvb, (int)NW);
  cvt_f32_to_bf16<<<(int)(NW / 256), 256, 0, stream>>>(Wo, wob, (int)NW);

  // Stage 2: LoRA left factors (fp32 for accuracy).
  lora_xa_kernel<<<NROW / 8, 256, 0, stream>>>(q, Aq, xaq);
  lora_xa_kernel<<<NROW / 8, 256, 0, stream>>>(k, Ak, xak);
  lora_xa_kernel<<<NROW / 8, 256, 0, stream>>>(v, Av, xav);

  // Stage 3: Q/K/V projections (+LoRA).  V stored transposed [B,H,64,T].
  dim3 gGemm(NROW / 128, D_MODEL / 128);
  gemm_wmma_kernel<<<gGemm, 256, 0, stream>>>(xqb, wqb, bq, xaq, Bq,
                                              nullptr, Qb,  1, 1);
  gemm_wmma_kernel<<<gGemm, 256, 0, stream>>>(xkb, wkb, bk, xak, Bk,
                                              nullptr, Kb,  1, 1);
  gemm_wmma_kernel<<<gGemm, 256, 0, stream>>>(xvb, wvb, bv, xav, Bv,
                                              nullptr, Vtb, 2, 1);

  // Stage 4: attention (32 head-batches x 16 query-tile blocks).
  dim3 gAttn(BATCH * NUM_HEADS, SEQ_T / 128);
  flash_attn_kernel<<<gAttn, 256, 0, stream>>>(Qb, Kb, Vtb, mask, ctx);

  // Stage 5: output projection (fp32 result).
  gemm_wmma_kernel<<<gGemm, 256, 0, stream>>>(ctx, wob, bo, nullptr, nullptr,
                                              out, nullptr, 0, 0);
}
